// Attention_49366354100559
// MI455X (gfx1250) — matrix operand order repaired and checked by reading, not compiled, not run on hardware
//
#include <hip/hip_runtime.h>
#include <stdint.h>

// ---------------------------------------------------------------------------
// CDNA5 (gfx1250) wave32 WMMA attention block.
// Fixed problem: B=2, T=2048, D=768, H=12, Dh=64.
// ---------------------------------------------------------------------------

typedef __bf16 v16bf __attribute__((ext_vector_type(16)));
typedef float  v8f   __attribute__((ext_vector_type(8)));

__device__ __forceinline__ v8f wmma_bf16(v16bf a, v16bf b, v8f c) {
    // emits v_wmma_f32_16x16x32_bf16
    return __builtin_amdgcn_wmma_f32_16x16x32_bf16(
        /*neg_a=*/false, a, /*neg_b=*/false, b,
        /*c_mod=*/(short)0, c, /*reuse_a=*/false, /*reuse_b=*/false);
}

// Low 32 bits of a generic pointer to LDS == LDS byte offset (aperture check
// in CDNA5 maps LDS flat addresses via addr[31:0]).
__device__ __forceinline__ uint32_t lds_off(const void* p) {
    return (uint32_t)(uintptr_t)p;
}

// Async global->LDS 16-byte copy, tracked by ASYNCcnt (no VGPR round-trip).
__device__ __forceinline__ void async_copy_b128(uint32_t lds_byte_off,
                                                const void* gptr) {
    asm volatile("global_load_async_to_lds_b128 %0, %1, off"
                 :: "v"(lds_byte_off), "v"((uint64_t)(uintptr_t)gptr)
                 : "memory");
}

__device__ __forceinline__ void wait_asynccnt0() {
    asm volatile("s_wait_asynccnt 0x0" ::: "memory");
}

// A fragment (16x32 bf16), rows striped over lanes, K over elements.
// lane<16: row=lane, K = half*16+j (j=0..15); lane>=16 same rows, K shifted by 16.
__device__ __forceinline__ v16bf a_frag(const __bf16* t, int stride, int lane) {
    const int half = lane >> 4, r = lane & 15;
    const __bf16* p = t + r * stride + half * 16;
    v16bf a;
#pragma unroll
    for (int j = 0; j < 8; ++j) { a[j] = p[j]; a[8 + j] = p[8 + j]; }
    return a;
}

// B fragment (32x16 bf16) from row-major [K][N] tile: element j <-> K = half*16+j,
// column = lane&15.
__device__ __forceinline__ v16bf b_frag(const __bf16* t, int stride, int lane) {
    const int half = lane >> 4, c = lane & 15;
    v16bf b;
#pragma unroll
    for (int j = 0; j < 16; ++j) b[j] = t[(half * 16 + j) * stride + c];
    return b;
}

// ---------------------------------------------------------------------------
// GEMM + bias: Y[M,N] = X[M,K] * W[K,N] + b[N].  X,W,b are f32; Y is OutT.
// Block tile 64x64, 128 threads (4 waves), wave w owns rows w*16..w*16+15.
// K-step 32 (one WMMA contraction). Next K-step tiles are prefetched into L2
// while the current WMMAs run.
// ---------------------------------------------------------------------------
template <typename OutT>
__global__ __launch_bounds__(128)
void gemm_bias(const float* __restrict__ X, const float* __restrict__ W,
               const float* __restrict__ bias, OutT* __restrict__ Y,
               int M, int N, int K) {
    constexpr int LDX = 40;  // 64 x 32 bf16 tile, padded
    constexpr int LDW = 72;  // 32 x 64 bf16 tile, padded
    __shared__ __bf16 ldsX[64 * LDX];
    __shared__ __bf16 ldsW[32 * LDW];

    const int tid    = threadIdx.x;
    const int lane   = tid & 31;
    const int wave   = tid >> 5;
    const int half   = lane >> 4;
    const int lanelo = lane & 15;
    const int m0     = blockIdx.x * 64;
    const int n0     = blockIdx.y * 64;

    v8f acc0 = {}, acc1 = {}, acc2 = {}, acc3 = {};

    for (int k0 = 0; k0 < K; k0 += 32) {
        __syncthreads();
        // X tile: 64 rows x 32 cols, f32 -> bf16  (512 float4 loads)
        for (int i = tid; i < 512; i += 128) {
            const int row = i >> 3, c4 = (i & 7) * 4;
            const float* src = &X[(size_t)(m0 + row) * K + k0 + c4];
            const float4 f = *(const float4*)src;
            if (k0 + 32 < K) __builtin_prefetch(src + 32, 0, 0);
            __bf16* dst = &ldsX[row * LDX + c4];
            dst[0] = (__bf16)f.x; dst[1] = (__bf16)f.y;
            dst[2] = (__bf16)f.z; dst[3] = (__bf16)f.w;
        }
        // W tile: 32 rows x 64 cols
        for (int i = tid; i < 512; i += 128) {
            const int kr = i >> 4, c4 = (i & 15) * 4;
            const float* src = &W[(size_t)(k0 + kr) * N + n0 + c4];
            const float4 f = *(const float4*)src;
            if (k0 + 32 < K) __builtin_prefetch(src + (size_t)32 * N, 0, 0);
            __bf16* dst = &ldsW[kr * LDW + c4];
            dst[0] = (__bf16)f.x; dst[1] = (__bf16)f.y;
            dst[2] = (__bf16)f.z; dst[3] = (__bf16)f.w;
        }
        __syncthreads();

        const v16bf a  = a_frag(ldsX + wave * 16 * LDX, LDX, lane);
        const v16bf b0 = b_frag(ldsW +  0, LDW, lane);
        const v16bf b1 = b_frag(ldsW + 16, LDW, lane);
        const v16bf b2 = b_frag(ldsW + 32, LDW, lane);
        const v16bf b3 = b_frag(ldsW + 48, LDW, lane);
        acc0 = wmma_bf16(a, b0, acc0);
        acc1 = wmma_bf16(a, b1, acc1);
        acc2 = wmma_bf16(a, b2, acc2);
        acc3 = wmma_bf16(a, b3, acc3);
    }

    // C/D layout: element r -> row = r + half*8, col = lane&15 (+16 per n-chunk)
#pragma unroll
    for (int r = 0; r < 8; ++r) {
        const int row = m0 + wave * 16 + r + half * 8;
        OutT* yp = Y + (size_t)row * N + n0;
        yp[ 0 + lanelo] = (OutT)(acc0[r] + bias[n0 +  0 + lanelo]);
        yp[16 + lanelo] = (OutT)(acc1[r] + bias[n0 + 16 + lanelo]);
        yp[32 + lanelo] = (OutT)(acc2[r] + bias[n0 + 32 + lanelo]);
        yp[48 + lanelo] = (OutT)(acc3[r] + bias[n0 + 48 + lanelo]);
    }
}

// ---------------------------------------------------------------------------
// Attention core. Grid: (T/64, B*H). 128 threads = 4 waves; wave owns 16 query
// rows. Streams 32-key tiles with online softmax. K/V tiles are staged with
// ASYNCcnt-tracked global->LDS copies (no VGPR round-trip). Faithful to the
// reference: the mask ZEROS future scores (they still enter the softmax), so
// every key tile is processed.
// ---------------------------------------------------------------------------
__global__ __launch_bounds__(128)
void attn_kernel(const __bf16* __restrict__ Q, const __bf16* __restrict__ Kb,
                 const __bf16* __restrict__ Vb, float* __restrict__ O, int T) {
    constexpr int H = 12, Dh = 64, D = 768;
    constexpr int LKV = 72;  // 32 x 64 bf16 K/V tiles; row = 144 B (16B aligned)
    constexpr int LP  = 40;  // 16 x 32 bf16 P staging, padded
    __shared__ __bf16 ldsK[32 * LKV];
    __shared__ __bf16 ldsV[32 * LKV];
    __shared__ __bf16 ldsP[4][16 * LP];

    const int tid    = threadIdx.x;
    const int lane   = tid & 31;
    const int wave   = tid >> 5;
    const int half   = lane >> 4;
    const int lanelo = lane & 15;

    const int bh = blockIdx.y;
    const int b  = bh / H, h = bh % H;
    const int q0 = blockIdx.x * 64 + wave * 16;

    const __bf16* qbase = Q  + (size_t)b * T * D + h * Dh;
    const __bf16* kbase = Kb + (size_t)b * T * D + h * Dh;
    const __bf16* vbase = Vb + (size_t)b * T * D + h * Dh;

    // Q A-fragments (16 rows x 64 dims = 2 fragments), loaded once.
    v16bf aq0, aq1;
    {
        const __bf16* qp = qbase + (size_t)(q0 + lanelo) * D;
#pragma unroll
        for (int j = 0; j < 8; ++j) {
            aq0[j]     = qp[      half * 16 + j];
            aq0[8 + j] = qp[8 +   half * 16 + j];
            aq1[j]     = qp[32 +  half * 16 + j];
            aq1[8 + j] = qp[40 +  half * 16 + j];
        }
    }

    v8f o0 = {}, o1 = {}, o2 = {}, o3 = {};
    float m[8], l[8];
#pragma unroll
    for (int r = 0; r < 8; ++r) { m[r] = -3.0e38f; l[r] = 0.0f; }

    for (int kt = 0; kt < T / 32; ++kt) {
        __syncthreads();  // previous tile fully consumed
        // Async K/V tile staging: 32 keys x 64 dims bf16 = 256 x 16B chunks
        // per tile; each of the 128 threads issues 2 chunks per tile.
#pragma unroll
        for (int i = tid; i < 256; i += 128) {
            const int key = i >> 3, c = (i & 7) * 8;   // c in bf16 elements
            const size_t goff = (size_t)(kt * 32 + key) * D + c;
            async_copy_b128(lds_off(&ldsK[key * LKV + c]), kbase + goff);
            async_copy_b128(lds_off(&ldsV[key * LKV + c]), vbase + goff);
        }
        wait_asynccnt0();
        __syncthreads();  // tiles visible to all waves

        // S = Q * K^T for 16 queries x 32 keys (two 16x16 tiles).
        v8f s0 = {}, s1 = {};
        {
            v16bf bk;
#pragma unroll
            for (int j = 0; j < 16; ++j) bk[j] = ldsK[lanelo * LKV + half * 16 + j];
            s0 = wmma_bf16(aq0, bk, s0);
#pragma unroll
            for (int j = 0; j < 16; ++j) bk[j] = ldsK[lanelo * LKV + 32 + half * 16 + j];
            s0 = wmma_bf16(aq1, bk, s0);
#pragma unroll
            for (int j = 0; j < 16; ++j) bk[j] = ldsK[(16 + lanelo) * LKV + half * 16 + j];
            s1 = wmma_bf16(aq0, bk, s1);
#pragma unroll
            for (int j = 0; j < 16; ++j) bk[j] = ldsK[(16 + lanelo) * LKV + 32 + half * 16 + j];
            s1 = wmma_bf16(aq1, bk, s1);
        }

        // Mask(->0) + scale + online softmax. Row r+half*8 lives across one
        // 16-lane half -> reduce with xor masks 1,2,4,8.
        float p0v[8], p1v[8], alpha[8];
#pragma unroll
        for (int r = 0; r < 8; ++r) {
            const int trow = q0 + r + half * 8;
            const int kc0  = kt * 32 + lanelo;
            const int kc1  = kc0 + 16;
            float x0 = ((kc0 > trow) ? 0.0f : (float)s0[r]) * 0.125f;
            float x1 = ((kc1 > trow) ? 0.0f : (float)s1[r]) * 0.125f;
            float tm = fmaxf(x0, x1);
#pragma unroll
            for (int msk = 1; msk < 16; msk <<= 1)
                tm = fmaxf(tm, __shfl_xor(tm, msk, 32));
            const float mn = fmaxf(m[r], tm);
            const float a  = __expf(m[r] - mn);
            const float p0 = __expf(x0 - mn);
            const float p1 = __expf(x1 - mn);
            float rs = p0 + p1;
#pragma unroll
            for (int msk = 1; msk < 16; msk <<= 1)
                rs += __shfl_xor(rs, msk, 32);
            l[r] = l[r] * a + rs;
            m[r] = mn;
            alpha[r] = a;
            p0v[r] = p0; p1v[r] = p1;
        }
#pragma unroll
        for (int r = 0; r < 8; ++r) {
            o0[r] *= alpha[r]; o1[r] *= alpha[r];
            o2[r] *= alpha[r]; o3[r] *= alpha[r];
        }

        // C-layout -> A-layout for P via per-wave LDS staging.
        __bf16* pw = &ldsP[wave][0];
#pragma unroll
        for (int r = 0; r < 8; ++r) {
            pw[(r + half * 8) * LP +      lanelo] = (__bf16)p0v[r];
            pw[(r + half * 8) * LP + 16 + lanelo] = (__bf16)p1v[r];
        }
        const v16bf ap = a_frag(pw, LP, lane);

        // O += P * V  (4 d-chunks of 16)
        o0 = wmma_bf16(ap, b_frag(ldsV +  0, LKV, lane), o0);
        o1 = wmma_bf16(ap, b_frag(ldsV + 16, LKV, lane), o1);
        o2 = wmma_bf16(ap, b_frag(ldsV + 32, LKV, lane), o2);
        o3 = wmma_bf16(ap, b_frag(ldsV + 48, LKV, lane), o3);
    }

    // Normalize and store into [B,T,D] layout (head h owns cols h*64..h*64+63).
#pragma unroll
    for (int r = 0; r < 8; ++r) {
        const float inv = 1.0f / l[r];
        const int trow  = q0 + r + half * 8;
        float* op = O + (size_t)(b * T + trow) * D + h * Dh;
        op[ 0 + lanelo] = o0[r] * inv;
        op[16 + lanelo] = o1[r] * inv;
        op[32 + lanelo] = o2[r] * inv;
        op[48 + lanelo] = o3[r] * inv;
    }
}

// ---------------------------------------------------------------------------
// Launch: QKV proj (bf16 out) -> attention -> output proj (f32 out).
// ---------------------------------------------------------------------------
extern "C" void kernel_launch(void* const* d_in, const int* in_sizes, int n_in,
                              void* d_out, int out_size, void* d_ws, size_t ws_size,
                              hipStream_t stream) {
    (void)in_sizes; (void)n_in; (void)out_size; (void)ws_size;
    const float* x  = (const float*)d_in[0];
    const float* Wq = (const float*)d_in[1];
    const float* bq = (const float*)d_in[2];
    const float* Wk = (const float*)d_in[3];
    const float* bk = (const float*)d_in[4];
    const float* Wv = (const float*)d_in[5];
    const float* bv = (const float*)d_in[6];
    const float* Wo = (const float*)d_in[7];
    const float* bo = (const float*)d_in[8];

    constexpr int B = 2, T = 2048, D = 768, H = 12;
    constexpr int M = B * T;

    char* ws = (char*)d_ws;
    __bf16* q  = (__bf16*)(ws);
    __bf16* k  = (__bf16*)(ws + (size_t)M * D * 2);
    __bf16* v  = (__bf16*)(ws + (size_t)M * D * 4);
    float*  ao = (float*)(ws + (size_t)M * D * 6);

    const dim3 gg(M / 64, D / 64);
    gemm_bias<__bf16><<<gg, 128, 0, stream>>>(x, Wq, bq, q, M, D, D);
    gemm_bias<__bf16><<<gg, 128, 0, stream>>>(x, Wk, bk, k, M, D, D);
    gemm_bias<__bf16><<<gg, 128, 0, stream>>>(x, Wv, bv, v, M, D, D);

    attn_kernel<<<dim3(T / 64, B * H), 128, 0, stream>>>(q, k, v, ao, T);

    gemm_bias<float><<<gg, 128, 0, stream>>>(ao, Wo, bo, (float*)d_out, M, D, D);
}
